// GCN_38147899523157
// MI455X (gfx1250) — matrix operand order repaired and checked by reading, not compiled, not run on hardware
//
#include <hip/hip_runtime.h>

typedef __attribute__((ext_vector_type(8)))  _Float16 v8h;
typedef __attribute__((ext_vector_type(16))) _Float16 v16h;
typedef __attribute__((ext_vector_type(8)))  float    v8f;

#define NNODE 25
#define HID   128
#define GPW   16               // graphs per workgroup
#define ROWS  (GPW * NNODE)    // 400 rows = 25 tiles of 16
#define MAXD  8                // max in-degree (actual max is 5)
#define NTHREADS 256           // 8 waves (wave32)

struct SharedMem {
    _Float16 h1[ROWS * HID];      // 100 KB  layer-1 activations (f16)
    _Float16 w2t[HID * HID];      //  32 KB  W2 transposed: w2t[n*128+k] = W2[k][n]
    float    xs[ROWS * 3];        // input features tile
    float    aggx[ROWS * 3];      // layer-1 aggregated features
    float    pooled[GPW * HID];   // per-graph pooled sums
    float    w1[3 * HID];
    float    b1s[HID];
    float    b2s[HID];
    float    wfc[HID * 2];
    float    bfc2[2];
    float    csr_c[NNODE * MAXD]; // folded edge coeff cs[src]*ci[dst]
    int      csr_s[NNODE * MAXD]; // neighbor (src) node id
    int      csr_n[NNODE];        // per-dst neighbor count
    float    cs[NNODE];
    float    ci[NNODE];
};

__global__ __launch_bounds__(NTHREADS, 1)
void gcn_fused(const float* __restrict__ x,  const int* __restrict__ src,
               const int* __restrict__ dst,  const float* __restrict__ W1,
               const float* __restrict__ b1, const float* __restrict__ W2,
               const float* __restrict__ b2, const float* __restrict__ Wfc,
               const float* __restrict__ bfc, float* __restrict__ out, int E)
{
    __shared__ SharedMem sm;
    const int tid = threadIdx.x;
    const int g0  = blockIdx.x * GPW;   // first graph of this block

    // ---- stage weights / biases into LDS (W2 transposed, cast to f16) ----
    for (int e = tid; e < HID * HID; e += NTHREADS) {
        int k = e >> 7, n = e & 127;
        sm.w2t[n * HID + k] = (_Float16)W2[e];
    }
    for (int e = tid; e < 3 * HID; e += NTHREADS) sm.w1[e] = W1[e];
    for (int e = tid; e < HID; e += NTHREADS) { sm.b1s[e] = b1[e]; sm.b2s[e] = b2[e]; }
    for (int e = tid; e < HID * 2; e += NTHREADS) sm.wfc[e] = Wfc[e];
    if (tid < 2) sm.bfc2[tid] = bfc[tid];
    for (int e = tid; e < GPW * HID; e += NTHREADS) sm.pooled[e] = 0.0f;

    // ---- degrees -> symmetric norm coefficients ----
    if (tid < NNODE) {
        int dout = 0, din = 0;
        for (int e = 0; e < E; ++e) { dout += (src[e] == tid); din += (dst[e] == tid); }
        sm.cs[tid] = dout > 0 ? rsqrtf((float)dout) : 0.0f;
        sm.ci[tid] = din  > 0 ? rsqrtf((float)din)  : 0.0f;
    }
    // ---- input tile ----
    for (int e = tid; e < ROWS * 3; e += NTHREADS) sm.xs[e] = x[g0 * (NNODE * 3) + e];
    __syncthreads();

    // ---- CSR by destination, fold cs*ci into edge coefficient ----
    if (tid < NNODE) {
        int cnt = 0;
        for (int e = 0; e < E; ++e) {
            if (dst[e] == tid && cnt < MAXD) {
                int s = src[e];
                sm.csr_s[tid * MAXD + cnt] = s;
                sm.csr_c[tid * MAXD + cnt] = sm.cs[s] * sm.ci[tid];
                ++cnt;
            }
        }
        sm.csr_n[tid] = cnt;
    }
    __syncthreads();

    // ---- layer 1 aggregation (3 features) ----
    for (int r = tid; r < ROWS; r += NTHREADS) {
        int n = r % NNODE;
        int gb = (r - n) * 3;              // (g*25)*3
        float a0 = 0.f, a1 = 0.f, a2 = 0.f;
        int cnt = sm.csr_n[n];
        for (int j = 0; j < cnt; ++j) {
            int   s = sm.csr_s[n * MAXD + j];
            float c = sm.csr_c[n * MAXD + j];
            const float* xp = &sm.xs[gb + s * 3];
            a0 += c * xp[0]; a1 += c * xp[1]; a2 += c * xp[2];
        }
        sm.aggx[r * 3 + 0] = a0; sm.aggx[r * 3 + 1] = a1; sm.aggx[r * 3 + 2] = a2;
    }
    __syncthreads();

    // ---- layer 1: [400,3] @ [3,128] + b1, ReLU -> h1 (f16 LDS) ----
    for (int e = tid; e < ROWS * HID; e += NTHREADS) {
        int r = e >> 7, c = e & 127;
        float v = sm.b1s[c]
                + sm.aggx[r * 3 + 0] * sm.w1[0 * HID + c]
                + sm.aggx[r * 3 + 1] * sm.w1[1 * HID + c]
                + sm.aggx[r * 3 + 2] * sm.w1[2 * HID + c];
        v = v > 0.f ? v : 0.f;
        sm.h1[e] = (_Float16)v;
    }
    __syncthreads();

    // ---- layer 2: per-wave 16-row tiles; A built in registers, WMMA GEMM ----
    const int wave = tid >> 5, lane = tid & 31;
    const int lhi  = lane >> 4;      // lane half (0: K-lo runs, 1: K-hi runs)
    const int lm   = lane & 15;      // M row within tile (A/D) / N column (B/D)

    for (int rt = wave; rt < ROWS / 16; rt += NTHREADS / 32) {
        const int row   = rt * 16 + lm;     // this lane's A row (both lane halves)
        const int n     = row % NNODE;
        const int g     = row / NNODE;
        const int hbase = (g * NNODE) * HID;
        const int cnt   = sm.csr_n[n];

        // Build A fragments (aggregated layer-1 feats) for all 4 k-blocks.
        v16h afrag[4];
#pragma unroll
        for (int kbi = 0; kbi < 4; ++kbi) {
            const int k0 = kbi * 32 + lhi * 8;   // contiguous runs k0..k0+7, k0+16..k0+23
            float acc[16];
#pragma unroll
            for (int i = 0; i < 16; ++i) acc[i] = 0.f;
            for (int j = 0; j < cnt; ++j) {
                int   s = sm.csr_s[n * MAXD + j];
                float c = sm.csr_c[n * MAXD + j];
                v8h h0 = *(const v8h*)&sm.h1[hbase + s * HID + k0];
                v8h h2 = *(const v8h*)&sm.h1[hbase + s * HID + k0 + 16];
#pragma unroll
                for (int i = 0; i < 8; ++i) {
                    acc[i]     += c * (float)h0[i];
                    acc[8 + i] += c * (float)h2[i];
                }
            }
            v16h a;
#pragma unroll
            for (int i = 0; i < 16; ++i) a[i] = (_Float16)acc[i];
            afrag[kbi] = a;
        }

        // 8 column tiles of the 128-wide output
        for (int ct = 0; ct < 8; ++ct) {
            const int col  = ct * 16 + lm;
            const float bias = sm.b2s[col];   // hoisted: overlaps the WMMA chain
            v8f d = {};
#pragma unroll
            for (int kbi = 0; kbi < 4; ++kbi) {
                const int k0 = kbi * 32 + lhi * 8;   // B: runs k0..k0+7, k0+16..k0+23 (match A)
                v8h blo = *(const v8h*)&sm.w2t[col * HID + k0];
                v8h bhi = *(const v8h*)&sm.w2t[col * HID + k0 + 16];
                v16h b;
#pragma unroll
                for (int i = 0; i < 8; ++i) { b[i] = blo[i]; b[8 + i] = bhi[i]; }
                d = __builtin_amdgcn_wmma_f32_16x16x32_f16(
                        false, afrag[kbi], false, b, (short)0, d, false, false);
            }
            // bias + ReLU + mean-pool accumulation (h2 never touches memory)
#pragma unroll
            for (int r8 = 0; r8 < 8; ++r8) {
                int m    = r8 + lhi * 8;
                int orow = rt * 16 + m;
                float v  = d[r8] + bias;
                v = v > 0.f ? v : 0.f;
                int og = orow / NNODE;
                atomicAdd(&sm.pooled[og * HID + col], v);
            }
        }
    }
    __syncthreads();

    // ---- final FC: pooled/25 @ Wfc + bfc, all 256 threads participate ----
    {
        int g    = tid >> 4;        // 0..15
        int c    = (tid >> 3) & 1;  // 0..1
        int part = tid & 7;         // 0..7 : 16-feature slice
        float s = 0.f;
        int f0 = part * 16;
#pragma unroll
        for (int f = 0; f < 16; ++f)
            s += sm.pooled[g * HID + f0 + f] * sm.wfc[(f0 + f) * 2 + c];
        // reduce across the 8 contiguous lanes of this (g,c) group (within wave32)
        s += __shfl_xor(s, 1, 32);
        s += __shfl_xor(s, 2, 32);
        s += __shfl_xor(s, 4, 32);
        if (part == 0)
            out[(g0 + g) * 2 + c] = sm.bfc2[c] + s * (1.0f / NNODE);
    }
}

extern "C" void kernel_launch(void* const* d_in, const int* in_sizes, int n_in,
                              void* d_out, int out_size, void* d_ws, size_t ws_size,
                              hipStream_t stream) {
    const float* x   = (const float*)d_in[0];
    const int*   src = (const int*)  d_in[1];
    const int*   dst = (const int*)  d_in[2];
    const float* W1  = (const float*)d_in[3];
    const float* b1  = (const float*)d_in[4];
    const float* W2  = (const float*)d_in[5];
    const float* b2  = (const float*)d_in[6];
    const float* Wfc = (const float*)d_in[7];
    const float* bfc = (const float*)d_in[8];
    float* out = (float*)d_out;

    const int Bn = in_sizes[0] / (NNODE * 3);   // 32768
    const int E  = in_sizes[1];                 // 49
    const int nblocks = Bn / GPW;               // 2048

    hipLaunchKernelGGL(gcn_fused, dim3(nblocks), dim3(NTHREADS), 0, stream,
                       x, src, dst, W1, b1, W2, b2, Wfc, bfc, out, E);
}
